// MolecularGraphConv_25838523252951
// MI455X (gfx1250) — matrix operand order repaired and checked by reading, compiled, not run on hardware
//
#include <hip/hip_runtime.h>
#include <hip/hip_bf16.h>
#include <stdint.h>

#define DEVFN __device__ __forceinline__

typedef __bf16 bf16_t;
typedef bf16_t v16bf __attribute__((ext_vector_type(16)));
typedef bf16_t v8bf  __attribute__((ext_vector_type(8)));
typedef float  v8f   __attribute__((ext_vector_type(8)));

constexpr int N_NODES  = 50000;
constexpr int N_EDGES  = 800000;
constexpr int NODE_DIM = 64;
constexpr int EDGE_DIM = 16;
constexpr int HIDDEN   = 128;
constexpr float EPS    = 1e-5f;

constexpr int ETPB = 16;  // 16-edge tiles per block  -> 3125 blocks
constexpr int NTPB = 5;   // 16-node tiles per block  -> 625 blocks

// ---------------- async global->LDS (gfx1250) with safe fallback -----------
#if defined(__gfx1250__) && __has_builtin(__builtin_amdgcn_global_load_async_to_lds_b128)
#define HAS_ASYNC_LDS 1
#define AS1 __attribute__((address_space(1)))
#define AS3 __attribute__((address_space(3)))
typedef int b128v __attribute__((__vector_size__(4 * sizeof(int))));
#endif

DEVFN void async_copy_b128(const unsigned short* src, unsigned short* dst) {
#ifdef HAS_ASYNC_LDS
  __builtin_amdgcn_global_load_async_to_lds_b128(
      (AS1 b128v*)src, (AS3 b128v*)dst, 0, 0);
#else
  *reinterpret_cast<v8bf*>(dst) = *reinterpret_cast<const v8bf*>(src);
#endif
}

DEVFN void wait_async() {
#ifdef HAS_ASYNC_LDS
#if __has_builtin(__builtin_amdgcn_s_wait_asynccnt)
  __builtin_amdgcn_s_wait_asynccnt(0);
#else
  asm volatile("s_wait_asynccnt 0x0" ::: "memory");
#endif
#endif
}

// ---------------- helpers ----------------

DEVFN unsigned short f2bf(float f) {
  union { float f; unsigned u; } v; v.f = f;
  unsigned r = v.u + 0x7FFFu + ((v.u >> 16) & 1u);  // round-to-nearest-even
  return (unsigned short)(r >> 16);
}

// A-fragment (16x32 bf16, MxK) from a row-major LDS tile.
DEVFN v16bf load_a_frag(const unsigned short* sA, int ld, int k0, int lane) {
  const int row  = lane & 15;
  const int koff = (lane >> 4) << 3;
  const unsigned short* base = sA + row * ld + k0 + koff;
  v8bf lo = *reinterpret_cast<const v8bf*>(base);
  v8bf hi = *reinterpret_cast<const v8bf*>(base + 16);
  v16bf a;
#pragma unroll
  for (int i = 0; i < 8; ++i) { a[i] = lo[i]; a[i + 8] = hi[i]; }
  return a;
}

// B-fragment from pre-packed weights (one 32B load per lane).
DEVFN v16bf load_b_frag(const unsigned short* Wp, int blk, int lane) {
  return *reinterpret_cast<const v16bf*>(Wp + ((size_t)blk * 32 + lane) * 16);
}

DEVFN v8f wmma_bf16(v16bf a, v16bf b, v8f c) {
  return __builtin_amdgcn_wmma_f32_16x16x32_bf16(false, a, false, b,
                                                 (short)0, c, false, false);
}

// ---------------- prep kernels ----------------

__global__ void pack_weights_kernel(const float* __restrict__ W, int K_in,
                                    unsigned short* __restrict__ dst) {
  const int blk  = blockIdx.x;   // kt*8 + nt
  const int lane = threadIdx.x;  // 0..31
  const int nt = blk & 7, kt = blk >> 3;
  const int n = nt * 16 + (lane & 15);
  const int kbase = kt * 32 + ((lane >> 4) << 3);
  unsigned short* d = dst + ((size_t)blk * 32 + lane) * 16;
#pragma unroll
  for (int e = 0; e < 16; ++e) {
    const int k = kbase + (e < 8 ? e : e + 8);
    const float v = (k < K_in) ? W[(size_t)k * HIDDEN + n] : 0.f;
    d[e] = f2bf(v);
  }
}

__global__ void cvt_bf16_kernel(const float* __restrict__ src,
                                unsigned short* __restrict__ dst, int count) {
  const int i = blockIdx.x * blockDim.x + threadIdx.x;
  if (i < count) dst[i] = f2bf(src[i]);
}

__global__ void zero_kernel(float* __restrict__ p, int count) {
  const int i = blockIdx.x * blockDim.x + threadIdx.x;
  if (i < count) p[i] = 0.f;
}

__global__ void finalize_stats_kernel(const float* __restrict__ stats,
                                      const float* __restrict__ gamma,
                                      const float* __restrict__ beta,
                                      float inv_cnt,
                                      float* __restrict__ scale,
                                      float* __restrict__ shift) {
  const int n = threadIdx.x;  // 128
  const float mean = stats[n] * inv_cnt;
  const float var  = stats[HIDDEN + n] * inv_cnt - mean * mean;
  const float rstd = rsqrtf(var + EPS);
  const float sc   = gamma[n] * rstd;
  scale[n] = sc;
  shift[n] = beta[n] - mean * sc;
}

// ---------------- edge pipeline ----------------
// A-tile layout per 16-edge tile: 16 rows x 160 bf16
//   cols [0,64)   = x[row]   (bf16, gathered)
//   cols [64,128) = x[col]   (bf16, gathered)
//   cols [128,144)= edge_attr (bf16, pre-converted)
//   cols [144,160)= zero padding (written once)
// Gather = 288 chunks of 16B per tile = 9 async b128 instrs (8 waves + wave0).
template <int PASS>
__global__ void __launch_bounds__(256)
edge_pass_kernel(const unsigned short* __restrict__ xbf,
                 const unsigned short* __restrict__ eabf,
                 const int* __restrict__ rowIdx,
                 const int* __restrict__ colIdx,
                 const unsigned short* __restrict__ We1p,
                 const float* __restrict__ be1,
                 float* __restrict__ statsE,
                 const float* __restrict__ scaleE,
                 const float* __restrict__ shiftE,
                 const unsigned short* __restrict__ We2p,
                 const float* __restrict__ be2,
                 const unsigned short* __restrict__ Wmp,
                 const float* __restrict__ bm,
                 float* __restrict__ agg) {
  __shared__ unsigned short sA[2][16 * 160];
  __shared__ unsigned short sB[16 * HIDDEN];
  __shared__ int   sRow[2][16];
  __shared__ float sStats[2 * HIDDEN];

  const int t = threadIdx.x;
  const int wave = t >> 5, lane = t & 31;
  const int ncol = wave * 16 + (lane & 15);
  const int mhi  = (lane >> 4) << 3;
  const size_t eblock = (size_t)blockIdx.x * (16 * ETPB);

  // one-time init: zero K padding (both buffers), zero stats scratch
  {
    const int r = t >> 4, c = 144 + (t & 15);
    sA[0][r * 160 + c] = 0;
    sA[1][r * 160 + c] = 0;
    if (PASS == 1) sStats[t] = 0.f;
  }

  // preload all B fragments for this wave's N-tile (kept in VGPRs)
  v16bf bW1[5];
#pragma unroll
  for (int kt = 0; kt < 5; ++kt) bW1[kt] = load_b_frag(We1p, kt * 8 + wave, lane);
  v16bf bW2[4], bW3[4];
  if (PASS == 2) {
#pragma unroll
    for (int kt = 0; kt < 4; ++kt) bW2[kt] = load_b_frag(We2p, kt * 8 + wave, lane);
#pragma unroll
    for (int kt = 0; kt < 4; ++kt) bW3[kt] = load_b_frag(Wmp, kt * 8 + wave, lane);
  }
  const float b1 = be1[ncol];
  float scE = 0.f, shE = 0.f, b2 = 0.f, b3 = 0.f;
  if (PASS == 2) { scE = scaleE[ncol]; shE = shiftE[ncol]; b2 = be2[ncol]; b3 = bm[ncol]; }

  // issue the 16B-chunk gather for one tile into buffer `buf`
  auto issue = [&](int buf, int tile) {
    const size_t ebase = eblock + (size_t)tile * 16;
    {
      const int c   = wave * 32 + lane;  // 0..255: x[row] then x[col]
      const int cc  = c & 127;
      const int row = cc >> 3, k = cc & 7;
      const size_t e = ebase + row;
      __builtin_prefetch((const void*)(rowIdx + ebase + 16), 0, 1);
      const int node = (c < 128) ? rowIdx[e] : colIdx[e];
      if (c < 128 && k == 0) sRow[buf][row] = node;
      const unsigned short* src = xbf + (size_t)node * NODE_DIM + k * 8;
      unsigned short* dst = &sA[buf][row * 160 + ((c < 128) ? 0 : 64) + k * 8];
      async_copy_b128(src, dst);
    }
    if (wave == 0) {  // edge_attr: 32 chunks
      const int row = lane >> 1, k = lane & 1;
      const size_t e = ebase + row;
      const unsigned short* src = eabf + e * EDGE_DIM + k * 8;
      unsigned short* dst = &sA[buf][row * 160 + 128 + k * 8];
      async_copy_b128(src, dst);
    }
  };

  issue(0, 0);
  float sAcc = 0.f, s2Acc = 0.f;

  for (int it = 0; it < ETPB; ++it) {
    wait_async();        // this wave's copies for tile `it` done
    __syncthreads();     // all waves' copies visible; prev compute finished
    if (it + 1 < ETPB) issue((it + 1) & 1, it + 1);  // overlaps compute below

    const unsigned short* A = sA[it & 1];
    const int* rowsP = sRow[it & 1];

    // GEMM1: [16x160] x [160x128]
    v8f acc = {};
#pragma unroll
    for (int kt = 0; kt < 5; ++kt)
      acc = wmma_bf16(load_a_frag(A, 160, kt * 32, lane), bW1[kt], acc);
#pragma unroll
    for (int r = 0; r < 8; ++r) acc[r] += b1;

    if (PASS == 1) {
#pragma unroll
      for (int r = 0; r < 8; ++r) { const float v = acc[r]; sAcc += v; s2Acc += v * v; }
      continue;
    }

    // BN + ReLU -> A2 (bf16 in LDS)
#pragma unroll
    for (int r = 0; r < 8; ++r) {
      const float v = fmaxf(acc[r] * scE + shE, 0.f);
      sB[(mhi + r) * HIDDEN + ncol] = f2bf(v);
    }
    __syncthreads();

    // GEMM2: edge_features = A2 @ We2 + be2
    v8f acc2 = {};
#pragma unroll
    for (int kt = 0; kt < 4; ++kt)
      acc2 = wmma_bf16(load_a_frag(sB, HIDDEN, kt * 32, lane), bW2[kt], acc2);
    __syncthreads();  // A2 fully consumed
#pragma unroll
    for (int r = 0; r < 8; ++r)
      sB[(mhi + r) * HIDDEN + ncol] = f2bf(acc2[r] + b2);
    __syncthreads();

    // GEMM3: messages = relu(A3 @ Wm + bm) -> scatter-add
    v8f acc3 = {};
#pragma unroll
    for (int kt = 0; kt < 4; ++kt)
      acc3 = wmma_bf16(load_a_frag(sB, HIDDEN, kt * 32, lane), bW3[kt], acc3);
#pragma unroll
    for (int r = 0; r < 8; ++r) {
      const float v = fmaxf(acc3[r] + b3, 0.f);
      atomicAdd(&agg[(size_t)rowsP[mhi + r] * HIDDEN + ncol], v);
    }
  }

  if (PASS == 1) {
    __syncthreads();
    atomicAdd(&sStats[ncol], sAcc);
    atomicAdd(&sStats[HIDDEN + ncol], s2Acc);
    __syncthreads();
    atomicAdd(&statsE[t], sStats[t]);
  }
}

// ---------------- node pipeline ----------------

__global__ void __launch_bounds__(256)
node_pass1_kernel(const unsigned short* __restrict__ xbf,
                  const float* __restrict__ agg,
                  const unsigned short* __restrict__ Wn1p,
                  const float* __restrict__ bn1,
                  float* __restrict__ statsN,
                  float* __restrict__ hn1) {
  __shared__ unsigned short sA[16 * 192];
  __shared__ float sStats[2 * HIDDEN];
  const int t = threadIdx.x;
  const int wave = t >> 5, lane = t & 31;
  const int ncol = wave * 16 + (lane & 15);
  const int mhi  = (lane >> 4) << 3;
  sStats[t] = 0.f;

  v16bf bW[6];
#pragma unroll
  for (int kt = 0; kt < 6; ++kt) bW[kt] = load_b_frag(Wn1p, kt * 8 + wave, lane);
  const float b1 = bn1[ncol];
  float sAcc = 0.f, s2Acc = 0.f;

  for (int it = 0; it < NTPB; ++it) {
    const size_t tile = (size_t)blockIdx.x * NTPB + it;
    {
      const int r = t >> 4, cg = t & 15;
      const size_t node = tile * 16 + r;
#pragma unroll
      for (int j = 0; j < 12; ++j) {
        const int c = cg + 16 * j;
        unsigned short v;
        if (c < 64) v = xbf[node * NODE_DIM + c];
        else        v = f2bf(agg[node * HIDDEN + (c - 64)]);
        sA[r * 192 + c] = v;
      }
    }
    __syncthreads();
    v8f acc = {};
#pragma unroll
    for (int kt = 0; kt < 6; ++kt)
      acc = wmma_bf16(load_a_frag(sA, 192, kt * 32, lane), bW[kt], acc);
#pragma unroll
    for (int r = 0; r < 8; ++r) {
      const float v = acc[r] + b1;
      sAcc += v; s2Acc += v * v;
      hn1[(tile * 16 + mhi + r) * HIDDEN + ncol] = v;
    }
    __syncthreads();  // before next fill overwrites sA
  }
  atomicAdd(&sStats[ncol], sAcc);
  atomicAdd(&sStats[HIDDEN + ncol], s2Acc);
  __syncthreads();
  atomicAdd(&statsN[t], sStats[t]);
}

__global__ void __launch_bounds__(256)
node_pass2_kernel(const float* __restrict__ hn1,
                  const float* __restrict__ scaleN,
                  const float* __restrict__ shiftN,
                  const unsigned short* __restrict__ Wn2p,
                  const float* __restrict__ bn2,
                  float* __restrict__ out) {
  __shared__ unsigned short sA[16 * HIDDEN];
  const int t = threadIdx.x;
  const int wave = t >> 5, lane = t & 31;
  const int ncol = wave * 16 + (lane & 15);
  const int mhi  = (lane >> 4) << 3;

  v16bf bW[4];
#pragma unroll
  for (int kt = 0; kt < 4; ++kt) bW[kt] = load_b_frag(Wn2p, kt * 8 + wave, lane);
  const float b2 = bn2[ncol];

  for (int it = 0; it < NTPB; ++it) {
    const size_t tile = (size_t)blockIdx.x * NTPB + it;
    {
      const int r = t >> 4, cg = t & 15;
      const size_t node = tile * 16 + r;
#pragma unroll
      for (int j = 0; j < 8; ++j) {
        const int c = cg + 16 * j;
        float v = hn1[node * HIDDEN + c];
        v = fmaxf(v * scaleN[c] + shiftN[c], 0.f);
        sA[r * HIDDEN + c] = f2bf(v);
      }
    }
    __syncthreads();
    v8f acc = {};
#pragma unroll
    for (int kt = 0; kt < 4; ++kt)
      acc = wmma_bf16(load_a_frag(sA, HIDDEN, kt * 32, lane), bW[kt], acc);
#pragma unroll
    for (int r = 0; r < 8; ++r)
      out[(tile * 16 + mhi + r) * HIDDEN + ncol] = acc[r] + b2;
    __syncthreads();
  }
}

// ---------------- launcher ----------------
extern "C" void kernel_launch(void* const* d_in, const int* in_sizes, int n_in,
                              void* d_out, int out_size, void* d_ws, size_t ws_size,
                              hipStream_t stream) {
  (void)in_sizes; (void)n_in; (void)out_size; (void)ws_size;
  const float* x    = (const float*)d_in[0];
  const int*   eidx = (const int*)d_in[1];
  const float* ea   = (const float*)d_in[2];
  const float* We1  = (const float*)d_in[3];
  const float* be1  = (const float*)d_in[4];
  const float* ge   = (const float*)d_in[5];
  const float* bbe  = (const float*)d_in[6];
  const float* We2  = (const float*)d_in[7];
  const float* be2  = (const float*)d_in[8];
  const float* Wm   = (const float*)d_in[9];
  const float* bm   = (const float*)d_in[10];
  const float* Wn1  = (const float*)d_in[11];
  const float* bn1  = (const float*)d_in[12];
  const float* gn   = (const float*)d_in[13];
  const float* bbn  = (const float*)d_in[14];
  const float* Wn2  = (const float*)d_in[15];
  const float* bn2  = (const float*)d_in[16];
  const int* rowIdx = eidx;
  const int* colIdx = eidx + N_EDGES;

  char* ws = (char*)d_ws;
  unsigned short* We1p = (unsigned short*)(ws + 0);       // 5*8 blocks
  unsigned short* We2p = (unsigned short*)(ws + 40960);   // 4*8
  unsigned short* Wmp  = (unsigned short*)(ws + 73728);   // 4*8
  unsigned short* Wn1p = (unsigned short*)(ws + 106496);  // 6*8
  unsigned short* Wn2p = (unsigned short*)(ws + 155648);  // 4*8
  float* scaleE = (float*)(ws + 188416);
  float* shiftE = (float*)(ws + 188928);
  float* scaleN = (float*)(ws + 189440);
  float* shiftN = (float*)(ws + 189952);
  float* statsE = (float*)(ws + 190464);                   // 256 f
  float* statsN = (float*)(ws + 191488);                   // 256 f
  float* agg    = (float*)(ws + 192512);                   // 50000*128 f
  float* hn1    = (float*)(ws + 25792512);                 // 50000*128 f
  unsigned short* xbf  = (unsigned short*)(ws + 51392512); // 50000*64 bf16
  unsigned short* eabf = (unsigned short*)(ws + 57792512); // 800000*16 bf16

  // 1) prep
  pack_weights_kernel<<<5 * 8, 32, 0, stream>>>(We1, 144, We1p);
  pack_weights_kernel<<<4 * 8, 32, 0, stream>>>(We2, 128, We2p);
  pack_weights_kernel<<<4 * 8, 32, 0, stream>>>(Wm, 128, Wmp);
  pack_weights_kernel<<<6 * 8, 32, 0, stream>>>(Wn1, 192, Wn1p);
  pack_weights_kernel<<<4 * 8, 32, 0, stream>>>(Wn2, 128, Wn2p);
  cvt_bf16_kernel<<<(N_NODES * NODE_DIM + 255) / 256, 256, 0, stream>>>(
      x, xbf, N_NODES * NODE_DIM);
  cvt_bf16_kernel<<<(N_EDGES * EDGE_DIM + 255) / 256, 256, 0, stream>>>(
      ea, eabf, N_EDGES * EDGE_DIM);
  const int zcount = 512 + N_NODES * HIDDEN;  // statsE|statsN|agg contiguous
  zero_kernel<<<(zcount + 255) / 256, 256, 0, stream>>>(statsE, zcount);

  // 2) edge pass 1: BN statistics of GEMM1 output
  edge_pass_kernel<1><<<N_EDGES / (16 * ETPB), 256, 0, stream>>>(
      xbf, eabf, rowIdx, colIdx, We1p, be1, statsE,
      scaleE, shiftE, We2p, be2, Wmp, bm, agg);
  finalize_stats_kernel<<<1, HIDDEN, 0, stream>>>(
      statsE, ge, bbe, 1.f / (float)N_EDGES, scaleE, shiftE);

  // 3) edge pass 2: full edge MLP + message scatter-add
  edge_pass_kernel<2><<<N_EDGES / (16 * ETPB), 256, 0, stream>>>(
      xbf, eabf, rowIdx, colIdx, We1p, be1, statsE,
      scaleE, shiftE, We2p, be2, Wmp, bm, agg);

  // 4) node pass 1: GEMM1 + BN stats (stores pre-activation)
  node_pass1_kernel<<<N_NODES / (16 * NTPB), 256, 0, stream>>>(
      xbf, agg, Wn1p, bn1, statsN, hn1);
  finalize_stats_kernel<<<1, HIDDEN, 0, stream>>>(
      statsN, gn, bbn, 1.f / (float)N_NODES, scaleN, shiftN);

  // 5) node pass 2: BN+ReLU -> GEMM2 -> output
  node_pass2_kernel<<<N_NODES / (16 * NTPB), 256, 0, stream>>>(
      hn1, scaleN, shiftN, Wn2p, bn2, (float*)d_out);
}
